// MultiHeadCrossAttention_88115549045475
// MI455X (gfx1250) — matrix operand order repaired and checked by reading, not compiled, not run on hardware
//
#include <hip/hip_runtime.h>
#include <math.h>

typedef __attribute__((ext_vector_type(16))) __bf16 bf16x16;
typedef __attribute__((ext_vector_type(8)))  float  f32x8;
typedef __attribute__((ext_vector_type(8)))  unsigned short u16x8;

union BF16Frag { u16x8 h[2]; bf16x16 v; };

__device__ __forceinline__ unsigned short f2bf(float f) {
    unsigned int u = __float_as_uint(f);
    u += 0x7fffu + ((u >> 16) & 1u);          // round-to-nearest-even
    return (unsigned short)(u >> 16);
}

__device__ __forceinline__ f32x8 zero8() {
    f32x8 z;
#pragma unroll
    for (int i = 0; i < 8; ++i) z[i] = 0.0f;
    return z;
}

__device__ __forceinline__ f32x8 wmma_bf16(bf16x16 a, bf16x16 b, f32x8 c) {
    return __builtin_amdgcn_wmma_f32_16x16x32_bf16(false, a, false, b, (short)0, c,
                                                   false, false);
}

// A-fragment 16x32 bf16: lane holds row (lane&15); element halves at
// k0+kb8 and k0+16+kb8 (kb8 = (lane>=16)?8:0). Two 16-byte loads.
__device__ __forceinline__ bf16x16 load_a_frag(const unsigned short* row, int k0, int kb8) {
    BF16Frag f;
    f.h[0] = *(const u16x8*)(row + k0 + kb8);
    f.h[1] = *(const u16x8*)(row + k0 + 16 + kb8);
    return f.v;
}

// B-fragment 32x16 bf16 where B[k][n] = Mat[n][k]: lane holds col n=(lane&15),
// element halves at p+kb8 and p+16+kb8 (same k order as A). Two 16-byte loads.
__device__ __forceinline__ bf16x16 load_b_frag(const unsigned short* p, int kb8) {
    BF16Frag f;
    f.h[0] = *(const u16x8*)(p + kb8);
    f.h[1] = *(const u16x8*)(p + 16 + kb8);
    return f.v;
}

__global__ void f32_to_bf16_kernel(const float* __restrict__ in,
                                   unsigned short* __restrict__ out, int n) {
    int i = blockIdx.x * blockDim.x + threadIdx.x;
    if (i < n) out[i] = f2bf(in[i]);
}

// Unified projection GEMM: C[M][N] = A[M][1024] @ W[N][1024]^T + bias.
// Block = 256 threads = 8 waves arranged 4(M) x 2(N); each wave: 64x64 tile.
// Grid: (N/128, M/256).
// mode 0: Q proj  -> out_q bf16 [B*8+h][1024][128], scaled by qscale
// mode 1: KV proj -> out_k bf16 [B*8+h][2048][128], out_vt bf16 [B*8+h][128][2048]
// mode 2: O proj  -> out_f32 [M][1024]
__global__ __launch_bounds__(256) void proj_gemm_kernel(
    const unsigned short* __restrict__ A,
    const unsigned short* __restrict__ W,
    const float* __restrict__ bias,
    int mode, float qscale,
    unsigned short* __restrict__ out_q,
    unsigned short* __restrict__ out_k,
    unsigned short* __restrict__ out_vt,
    float* __restrict__ out_f32)
{
    const int lane = threadIdx.x & 31;
    const int wave = threadIdx.x >> 5;
    const int li   = lane & 15;
    const int hi   = lane >> 4;
    const int kb8  = hi * 8;
    const int kb16 = hi * 16;

    const int m0 = blockIdx.y * 256 + (wave >> 1) * 64;
    const int n0 = blockIdx.x * 128 + (wave & 1) * 64;

    const unsigned short* arow[4];
    const unsigned short* brow[4];
#pragma unroll
    for (int mi = 0; mi < 4; ++mi) arow[mi] = A + (size_t)(m0 + mi * 16 + li) * 1024;
#pragma unroll
    for (int nj = 0; nj < 4; ++nj) brow[nj] = W + (size_t)(n0 + nj * 16 + li) * 1024;

    f32x8 acc[4][4];
#pragma unroll
    for (int mi = 0; mi < 4; ++mi)
#pragma unroll
        for (int nj = 0; nj < 4; ++nj) acc[mi][nj] = zero8();

    for (int k0 = 0; k0 < 1024; k0 += 32) {
        bf16x16 af[4], bfm[4];
#pragma unroll
        for (int mi = 0; mi < 4; ++mi) af[mi] = load_a_frag(arow[mi], k0, kb8);
#pragma unroll
        for (int nj = 0; nj < 4; ++nj) bfm[nj] = load_b_frag(brow[nj] + k0, kb8);
#pragma unroll
        for (int mi = 0; mi < 4; ++mi)
#pragma unroll
            for (int nj = 0; nj < 4; ++nj)
                acc[mi][nj] = wmma_bf16(af[mi], bfm[nj], acc[mi][nj]);
    }

#pragma unroll
    for (int nj = 0; nj < 4; ++nj) {
        const int n = n0 + nj * 16 + li;
        const float bv = bias[n];
#pragma unroll
        for (int mi = 0; mi < 4; ++mi) {
#pragma unroll
            for (int r = 0; r < 8; ++r) {
                const int m = m0 + mi * 16 + 8 * hi + r;
                const float v = acc[mi][nj][r] + bv;
                if (mode == 0) {
                    const int b = m >> 10, ld = m & 1023;
                    const int h = n >> 7,  d  = n & 127;
                    out_q[((size_t)(b * 8 + h) * 1024 + ld) * 128 + d] = f2bf(v * qscale);
                } else if (mode == 1) {
                    const int b = m >> 11, le = m & 2047;
                    const int h = n >> 8,  c  = n & 255;
                    if (c < 128)
                        out_k[((size_t)(b * 8 + h) * 2048 + le) * 128 + c] = f2bf(v);
                    else
                        out_vt[((size_t)(b * 8 + h) * 128 + (c - 128)) * 2048 + le] = f2bf(v);
                } else {
                    out_f32[(size_t)m * 1024 + n] = v;
                }
            }
        }
    }
}

// Flash attention: one wave per 16 Q rows of one (b,h).
// Q pre-scaled by log2e/sqrt(128) so softmax uses exp2.
// Grid: 256 blocks x 256 threads = 2048 waves = B*H*(Ld/16).
__global__ __launch_bounds__(256) void attn_kernel(
    const unsigned short* __restrict__ Q,    // [BH][1024][128] bf16
    const unsigned short* __restrict__ K,    // [BH][2048][128] bf16
    const unsigned short* __restrict__ Vt,   // [BH][128][2048] bf16
    const float* __restrict__ mask,          // [B][1024][2048] f32
    unsigned short* __restrict__ vals)       // [B][1024][1024] bf16
{
    __shared__ alignas(16) unsigned short Plds[8][16 * 64];

    const int lane = threadIdx.x & 31;
    const int wave = threadIdx.x >> 5;
    const int li   = lane & 15;
    const int hi   = lane >> 4;
    const int kb8  = hi * 8;
    const int kb16 = hi * 16;

    const int wg = blockIdx.x * 8 + wave;
    const int bh = wg >> 6;        // 0..31  (b*8+h)
    const int qt = wg & 63;        // q tile index, 16 rows each
    const int b  = bh >> 3;
    const int h  = bh & 7;

    const unsigned short* Qb = Q  + (size_t)bh * 1024 * 128;
    const unsigned short* Kb = K  + (size_t)bh * 2048 * 128;
    const unsigned short* Vb = Vt + (size_t)bh * 128  * 2048;

    bf16x16 qf[4];
    {
        const unsigned short* qrow = Qb + (size_t)(qt * 16 + li) * 128;
#pragma unroll
        for (int t = 0; t < 4; ++t) qf[t] = load_a_frag(qrow, t * 32, kb8);
    }

    f32x8 of[8];
#pragma unroll
    for (int dj = 0; dj < 8; ++dj) of[dj] = zero8();

    float run_m[8], run_s[8];
#pragma unroll
    for (int r = 0; r < 8; ++r) { run_m[r] = -1e30f; run_s[r] = 0.0f; }

    const float* mrow[8];
#pragma unroll
    for (int r = 0; r < 8; ++r)
        mrow[r] = mask + ((size_t)b * 1024 + qt * 16 + 8 * hi + r) * 2048;

    const float LOG2E = 1.4426950408889634f;

    for (int nc = 0; nc < 2048; nc += 64) {
        // S tile 16x64 = 4 frags; contraction over d=128 in 4 WMMA k-steps
        f32x8 s[4];
#pragma unroll
        for (int j = 0; j < 4; ++j) {
            f32x8 a = zero8();
            const int kcol = nc + j * 16 + li;
            const unsigned short* kp = Kb + (size_t)kcol * 128;
#pragma unroll
            for (int t = 0; t < 4; ++t)
                a = wmma_bf16(qf[t], load_b_frag(kp + t * 32, kb8), a);
#pragma unroll
            for (int r = 0; r < 8; ++r)
                a[r] += LOG2E * mrow[r][kcol];
            s[j] = a;
        }

        // per-row chunk max; rows live across the 16 lanes of each half-wave
        float alpha[8];
#pragma unroll
        for (int r = 0; r < 8; ++r) {
            float v = s[0][r];
#pragma unroll
            for (int j = 1; j < 4; ++j) v = fmaxf(v, s[j][r]);
#pragma unroll
            for (int d = 1; d < 16; d <<= 1) v = fmaxf(v, __shfl_xor(v, d));
            const float nm = fmaxf(run_m[r], v);
            alpha[r] = exp2f(run_m[r] - nm);
            run_m[r] = nm;
        }

        float cs[8];
#pragma unroll
        for (int r = 0; r < 8; ++r) cs[r] = 0.0f;
#pragma unroll
        for (int j = 0; j < 4; ++j)
#pragma unroll
            for (int r = 0; r < 8; ++r) {
                const float p = exp2f(s[j][r] - run_m[r]);
                s[j][r] = p;
                cs[r] += p;
            }
#pragma unroll
        for (int r = 0; r < 8; ++r) {
            float v = cs[r];
#pragma unroll
            for (int d = 1; d < 16; d <<= 1) v += __shfl_xor(v, d);
            run_s[r] = run_s[r] * alpha[r] + v;
        }
#pragma unroll
        for (int dj = 0; dj < 8; ++dj)
#pragma unroll
            for (int r = 0; r < 8; ++r) of[dj][r] *= alpha[r];

        // transpose P (C layout -> A layout) through wave-private LDS slice
        unsigned short* P = &Plds[wave][0];
#pragma unroll
        for (int j = 0; j < 4; ++j)
#pragma unroll
            for (int r = 0; r < 8; ++r)
                P[(8 * hi + r) * 64 + j * 16 + li] = f2bf(s[j][r]);

        bf16x16 pa[2];
#pragma unroll
        for (int t2 = 0; t2 < 2; ++t2) {
            BF16Frag f;
            const unsigned short* pr = P + li * 64 + t2 * 32;
            f.h[0] = *(const u16x8*)(pr + kb8);
            f.h[1] = *(const u16x8*)(pr + 16 + kb8);
            pa[t2] = f.v;
        }

        // O += P @ V : contraction over 64 chunk columns (2 WMMA k-steps)
#pragma unroll
        for (int dj = 0; dj < 8; ++dj) {
            const unsigned short* vp = Vb + (size_t)(dj * 16 + li) * 2048 + nc;
#pragma unroll
            for (int t2 = 0; t2 < 2; ++t2)
                of[dj] = wmma_bf16(pa[t2], load_b_frag(vp + t2 * 32, kb8), of[dj]);
        }
    }

    // normalize and write vals[b][row][h*128 + d] as bf16
#pragma unroll
    for (int r = 0; r < 8; ++r) {
        const float inv = 1.0f / run_s[r];
        const int row = qt * 16 + 8 * hi + r;
#pragma unroll
        for (int dj = 0; dj < 8; ++dj)
            vals[((size_t)(b * 1024 + row)) * 1024 + h * 128 + dj * 16 + li] =
                f2bf(of[dj][r] * inv);
    }
}

extern "C" void kernel_launch(void* const* d_in, const int* in_sizes, int n_in,
                              void* d_out, int out_size, void* d_ws, size_t ws_size,
                              hipStream_t stream) {
    (void)in_sizes; (void)n_in; (void)out_size; (void)ws_size;
    const float* x    = (const float*)d_in[0];
    const float* enc  = (const float*)d_in[1];
    const float* mask = (const float*)d_in[2];
    const float* Wkv  = (const float*)d_in[3];
    const float* bkv  = (const float*)d_in[4];
    const float* Wq   = (const float*)d_in[5];
    const float* bq   = (const float*)d_in[6];
    const float* Wo   = (const float*)d_in[7];
    const float* bo   = (const float*)d_in[8];
    float* out = (float*)d_out;

    char* ws = (char*)d_ws;
    size_t off = 0;
    auto alloc = [&](size_t bytes) {
        void* p = ws + off;
        off += (bytes + 255) & ~(size_t)255;
        return p;
    };
    unsigned short* Xbf   = (unsigned short*)alloc((size_t)4 * 1024 * 1024 * 2);
    unsigned short* Encbf = (unsigned short*)alloc((size_t)4 * 2048 * 1024 * 2);
    unsigned short* Wqbf  = (unsigned short*)alloc((size_t)1024 * 1024 * 2);
    unsigned short* Wkvbf = (unsigned short*)alloc((size_t)2048 * 1024 * 2);
    unsigned short* Wobf  = (unsigned short*)alloc((size_t)1024 * 1024 * 2);
    unsigned short* Qbuf  = (unsigned short*)alloc((size_t)32 * 1024 * 128 * 2);
    unsigned short* Kbuf  = (unsigned short*)alloc((size_t)32 * 2048 * 128 * 2);
    unsigned short* Vtbuf = (unsigned short*)alloc((size_t)32 * 128 * 2048 * 2);
    unsigned short* Vals  = (unsigned short*)alloc((size_t)4 * 1024 * 1024 * 2);

    auto cvt = [&](const float* src, unsigned short* dst, int n) {
        f32_to_bf16_kernel<<<(n + 255) / 256, 256, 0, stream>>>(src, dst, n);
    };
    cvt(x,   Xbf,   4 * 1024 * 1024);
    cvt(enc, Encbf, 4 * 2048 * 1024);
    cvt(Wq,  Wqbf,  1024 * 1024);
    cvt(Wkv, Wkvbf, 2048 * 1024);
    cvt(Wo,  Wobf,  1024 * 1024);

    const float qscale = 1.4426950408889634f / sqrtf(128.0f);

    // KV projection: M = 4*2048 = 8192, N = 2048
    proj_gemm_kernel<<<dim3(2048 / 128, 8192 / 256), 256, 0, stream>>>(
        Encbf, Wkvbf, bkv, 1, 0.0f, nullptr, Kbuf, Vtbuf, nullptr);
    // Q projection: M = 4*1024 = 4096, N = 1024
    proj_gemm_kernel<<<dim3(1024 / 128, 4096 / 256), 256, 0, stream>>>(
        Xbf, Wqbf, bq, 0, qscale, Qbuf, nullptr, nullptr, nullptr);
    // attention: B*H*(Ld/16) = 2048 waves -> 256 blocks of 8 waves
    attn_kernel<<<256, 256, 0, stream>>>(Qbuf, Kbuf, Vtbuf, mask, Vals);
    // output projection: M = 4096, N = 1024, f32 result
    proj_gemm_kernel<<<dim3(1024 / 128, 4096 / 256), 256, 0, stream>>>(
        Vals, Wobf, bo, 2, 0.0f, nullptr, nullptr, nullptr, out);
}
